// HRNN_encoder_4140348474010
// MI455X (gfx1250) — matrix operand order repaired and checked by reading, not compiled, not run on hardware
//
#include <hip/hip_runtime.h>
#include <math.h>

// ---------------------------------------------------------------------------
// HRNN encoder for MI455X (gfx1250).
// GEMMs use V_WMMA_F32_16X16X32_BF16 (wave32). Big per-depth GEMM uses a
// 2x3-tile wave block (A frag reused 3x, B frag reused 2x, 192 WMMA/wave);
// the latency-critical per-step GEMM uses 1 tile/wave for max parallelism.
// W/U are pre-swizzled once into B-fragment bf16 layout; A operands are
// converted f32 -> bf16 on the fly (b128 loads + v_cvt_pk_bf16_f32).
// ---------------------------------------------------------------------------

typedef __bf16 bf16_t;
typedef bf16_t bf16x16 __attribute__((ext_vector_type(16)));
typedef float  f32x8   __attribute__((ext_vector_type(8)));

namespace {
constexpr int kB    = 128;   // batch
constexpr int kT    = 128;   // seq len
constexpr int kI    = 512;   // input dim
constexpr int kH    = 512;   // hidden dim
constexpr int kD    = 1024;  // I + H
constexpr int kTb   = 128;   // bucket size
constexpr int kDepth = 4;
constexpr int kHP1  = kH + 1;        // 513
constexpr int kNT   = 33;            // N tiles of 16 (528 >= 513)
constexpr int kNp   = kNT * 16;      // 528 padded N
constexpr int kKT   = kD / 32;       // 32 K-steps of 32
constexpr float kEPS = 1e-5f;
}

__device__ __forceinline__ float hsig(float x) {
  return fminf(fmaxf(0.2f * x + 0.5f, 0.f), 1.f);
}

// --- prep: build time-major padded input sequence (Tb, B, D) ---------------
__global__ void k_prep_seq(const float* __restrict__ x, float* __restrict__ seq) {
  size_t idx = (size_t)blockIdx.x * blockDim.x + threadIdx.x;
  if (idx >= (size_t)kTb * kB * kD) return;
  int j = (int)(idx % kD);
  size_t r = idx / kD;
  int b = (int)(r % kB);
  int t = (int)(r / kB);
  float v = 0.f;
  if (j < kI) v = x[((size_t)b * kT + t) * kI + j];
  seq[idx] = v;
}

// --- prep: fk/hv initial state + mask-derived m, m2 ------------------------
__global__ void k_prep_scal(const int* __restrict__ mask,
                            float* __restrict__ fkA, float* __restrict__ hvA,
                            unsigned char* __restrict__ mB,
                            unsigned char* __restrict__ m2B) {
  int idx = blockIdx.x * blockDim.x + threadIdx.x;
  if (idx >= kTb * kB) return;
  int b = idx % kB, t = idx / kB;
  fkA[idx] = 0.f;
  hvA[idx] = 1.f;
  unsigned char mt = (mask[(size_t)b * kT + t] > 0) ? 1 : 0;
  mB[idx] = mt;
  unsigned char mp = (t > 0) ? ((mask[(size_t)b * kT + (t - 1)] > 0) ? 1 : 0) : 0;
  m2B[idx] = (t > 0 && mp && !mt) ? 1 : 0;   // m2[t] = m[t-1] & ~m[t]
}

// --- pack W/U (D x 513 f32) into WMMA B-fragment bf16 layout ---------------
// frag[(kt*kNT + nt)*512 + lane*16 + j] = B[kt*32 + (lane>>4)*16 + j][nt*16 + lane&15]
__global__ void k_pack(const float* __restrict__ Wsrc, bf16_t* __restrict__ frag) {
  int idx = blockIdx.x * blockDim.x + threadIdx.x;
  if (idx >= kKT * kNT * 512) return;
  int f = idx >> 9;
  int e = idx & 511;
  int lane = e >> 4, j = e & 15;
  int kt = f / kNT, nt = f % kNT;
  int krow = kt * 32 + (lane >> 4) * 16 + j;
  int n = nt * 16 + (lane & 15);
  float v = (n < kHP1) ? Wsrc[(size_t)krow * kHP1 + n] : 0.f;
  frag[idx] = (bf16_t)v;
}

// --- WMMA GEMM: C = A(f32, stride kD) x Bfrag(bf16), C stride kNp ----------
// Wave computes an MTG x NTG block of 16x16 tiles. kNT must be divisible by NTG.
template <int MTG, int NTG>
__global__ void k_gemm(const float* __restrict__ A, const bf16_t* __restrict__ Bf,
                       float* __restrict__ C, int MGroups) {
  constexpr int NGroups = kNT / NTG;
  int wv = (int)((blockIdx.x * blockDim.x + threadIdx.x) >> 5);
  int lane = threadIdx.x & 31;
  int total = MGroups * NGroups;
  if (wv >= total) return;            // wave-uniform exit (EXEC all-1 inside)
  int mg = wv / NGroups, ng = wv % NGroups;
  int half = lane >> 4;
  int rbase = mg * (16 * MTG) + (lane & 15);

  f32x8 acc[MTG][NTG];
#pragma unroll
  for (int mi = 0; mi < MTG; ++mi)
#pragma unroll
    for (int ni = 0; ni < NTG; ++ni)
      acc[mi][ni] = (f32x8){0.f, 0.f, 0.f, 0.f, 0.f, 0.f, 0.f, 0.f};

  for (int kt = 0; kt < kKT; ++kt) {
    bf16x16 a[MTG];
#pragma unroll
    for (int mi = 0; mi < MTG; ++mi) {
      const float* p0 = A + (size_t)(rbase + mi * 16) * kD + kt * 32 + half * 16;
      const float* p1 = p0 + 8;
#pragma unroll
      for (int e = 0; e < 8; ++e) a[mi][e] = (bf16_t)p0[e];
#pragma unroll
      for (int e = 0; e < 8; ++e) a[mi][8 + e] = (bf16_t)p1[e];
      if (kt + 1 < kKT) __builtin_prefetch(p0 + 32, 0, 1);  // next K-slab
    }
#pragma unroll
    for (int ni = 0; ni < NTG; ++ni) {
      bf16x16 bfr = *(const bf16x16*)(
          Bf + ((size_t)(kt * kNT + ng * NTG + ni) * 512 + lane * 16));
#pragma unroll
      for (int mi = 0; mi < MTG; ++mi)
        acc[mi][ni] = __builtin_amdgcn_wmma_f32_16x16x32_bf16(
            false, a[mi], false, bfr, (short)0, acc[mi][ni], false, false);
    }
  }

  int n = lane & 15;
#pragma unroll
  for (int mi = 0; mi < MTG; ++mi)
#pragma unroll
    for (int ni = 0; ni < NTG; ++ni)
#pragma unroll
      for (int v = 0; v < 8; ++v)
        C[(size_t)(mg * 16 * MTG + mi * 16 + half * 8 + v) * kNp +
          (ng * NTG + ni) * 16 + n] = acc[mi][ni][v];
}

// --- row LayerNorm in place over first 513 of 528 columns ------------------
__global__ void k_ln_rows(float* __restrict__ S, const float* __restrict__ g,
                          const float* __restrict__ bt) {
  __shared__ float rs[256], rq[256];
  int row = blockIdx.x;
  int tid = threadIdx.x;
  float* p = S + (size_t)row * kNp;
  float s = 0.f, q = 0.f;
  for (int j = tid; j < kHP1; j += 256) { float v = p[j]; s += v; q += v * v; }
  rs[tid] = s; rq[tid] = q;
  __syncthreads();
  for (int o = 128; o > 0; o >>= 1) {
    if (tid < o) { rs[tid] += rs[tid + o]; rq[tid] += rq[tid + o]; }
    __syncthreads();
  }
  float mean = rs[0] * (1.f / kHP1);
  float var  = fmaxf(rq[0] * (1.f / kHP1) - mean * mean, 0.f);
  float inv  = 1.f / (sqrtf(var + kEPS) + kEPS);
  for (int j = tid; j < kHP1; j += 256)
    p[j] = g[j] * ((p[j] - mean) * inv) + bt[j];
}

// --- fused per-step LN(h@U) + gate recurrence + state update ---------------
__global__ void k_step_update(const float* __restrict__ S0, const float* __restrict__ S2,
                              const float* __restrict__ xin, float* __restrict__ out,
                              const float* __restrict__ fk_prev, const float* __restrict__ hv_prev,
                              float* __restrict__ fk_cur, float* __restrict__ hv_cur,
                              const unsigned char* __restrict__ mB,
                              const unsigned char* __restrict__ m2B,
                              const float* __restrict__ bvec, const float* __restrict__ g1,
                              const float* __restrict__ b1, int t) {
  __shared__ float rs[256], rq[256];
  __shared__ float shv[3];
  __shared__ int shm;
  int b = blockIdx.x, tid = threadIdx.x;
  const float* raw = S2 + (size_t)b * kNp;          // h_{t-1} @ U (raw); zero at t==0
  float s = 0.f, q = 0.f;
  if (t > 0)
    for (int j = tid; j < kHP1; j += 256) { float v = raw[j]; s += v; q += v * v; }
  rs[tid] = s; rq[tid] = q;
  __syncthreads();
  for (int o = 128; o > 0; o >>= 1) {
    if (tid < o) { rs[tid] += rs[tid + o]; rq[tid] += rq[tid + o]; }
    __syncthreads();
  }
  float mean = rs[0] * (1.f / kHP1);
  float var  = fmaxf(rq[0] * (1.f / kHP1) - mean * mean, 0.f);
  float inv  = 1.f / (sqrtf(var + kEPS) + kEPS);

  if (tid == 0) {
    float raw0   = (t > 0) ? raw[0] : 0.f;
    float sum2_0 = g1[0] * ((raw0 - mean) * inv) + b1[0];
    float s_0    = S0[((size_t)t * kB + b) * kNp] + sum2_0 + bvec[0];
    float fk_both = hsig(s_0);
    float fk_t1   = hsig(sum2_0 + bvec[0]);
    float fk_tm1  = (t > 0) ? fk_cur[(t - 1) * kB + b] : 0.f;
    float hv_tm1  = (t > 0) ? hv_cur[(t - 1) * kB + b] : 0.f;
    float fkp_tm1 = fk_prev[t * kB + b];
    float fkp     = (t + 1 < kTb) ? fk_prev[(t + 1) * kB + b] : 0.f;  // shifted
    float hvp     = hv_prev[t * kB + b];
    int mk  = mB[t * kB + b];
    int mk2 = m2B[t * kB + b];
    float fk = fkp_tm1 + (1.f - fkp_tm1) * (fk_tm1 * fk_both + (1.f - fk_tm1) * fk_t1);
    if (mk2) fk = 0.f;
    float h_only = hv_tm1 * fk * (fkp + (1.f - fkp) * (1.f - hvp));
    float x_only = hvp * (1.f - fkp) * (1.f - fk + fk * (1.f - hv_tm1));
    float both   = (1.f - fkp) * fk * hv_tm1 * hvp;
    float hv = 1.f - (1.f - h_only) * (1.f - x_only) * (1.f - both);
    float fk_f = mk ? fk : fk_tm1;
    float hv_f = mk ? hv : hv_tm1;
    if (mk2) fk_f = 0.f;
    fk_cur[t * kB + b] = fk_f;
    hv_cur[t * kB + b] = hv_f;
    shv[0] = h_only; shv[1] = x_only; shv[2] = both;
    shm = mk;
  }
  __syncthreads();
  float h_only = shv[0], x_only = shv[1], both = shv[2];
  int mk = shm;
  const float* s0row = S0 + ((size_t)t * kB + b) * kNp;
  const float* xrow  = xin + ((size_t)t * kB + b) * kD;
  const float* hprev = out + ((size_t)(t > 0 ? t - 1 : 0) * kB + b) * kD;
  float* hrow        = out + ((size_t)t * kB + b) * kD;
  for (int j = tid; j < kD; j += 256) {
    float h_tm1v = (t > 0) ? hprev[j] : 0.f;
    float x_t = xrow[j];
    float h_new = 0.f;
    if (j >= kI) {
      int k = j - kI + 1;                              // s index 1..512
      float rawk  = (t > 0) ? raw[k] : 0.f;
      float sum2k = g1[k] * ((rawk - mean) * inv) + b1[k];
      float sk    = s0row[k] + sum2k + bvec[k];
      h_new = tanhf(sk);
    }
    float h = h_only * h_tm1v + x_only * x_t + both * h_new;
    if (!mk) h = h_tm1v;
    hrow[j] = h;
  }
}

// --- final slice: h_final[-1][:, I:] --------------------------------------
__global__ void k_final(const float* __restrict__ seq, float* __restrict__ o) {
  int idx = blockIdx.x * blockDim.x + threadIdx.x;
  if (idx >= kB * kH) return;
  int b = idx / kH, k = idx % kH;
  o[idx] = seq[((size_t)(kTb - 1) * kB + b) * kD + kI + k];
}

extern "C" void kernel_launch(void* const* d_in, const int* in_sizes, int n_in,
                              void* d_out, int out_size, void* d_ws, size_t ws_size,
                              hipStream_t stream) {
  (void)in_sizes; (void)n_in; (void)out_size; (void)ws_size;
  const float* x    = (const float*)d_in[0];
  const int*   mask = (const int*)d_in[1];
  const float* W    = (const float*)d_in[2];
  const float* U    = (const float*)d_in[3];
  const float* bvec = (const float*)d_in[4];
  const float* gam  = (const float*)d_in[5];
  const float* bet  = (const float*)d_in[6];
  const float* g0 = gam;        const float* g1 = gam + kHP1;
  const float* b0 = bet;        const float* b1 = bet + kHP1;

  char* ws = (char*)d_ws;
  size_t off = 0;
  auto take = [&](size_t bytes) -> char* {
    char* p = ws + off;
    off = (off + bytes + 255) & ~(size_t)255;
    return p;
  };
  float*  seqA = (float*)take(sizeof(float) * (size_t)kTb * kB * kD);   // 64 MB
  float*  seqB = (float*)take(sizeof(float) * (size_t)kTb * kB * kD);   // 64 MB
  float*  S0   = (float*)take(sizeof(float) * (size_t)kTb * kB * kNp);  // 34.6 MB
  float*  S2   = (float*)take(sizeof(float) * (size_t)kB * kNp);
  float*  fkA  = (float*)take(sizeof(float) * kTb * kB);
  float*  fkB  = (float*)take(sizeof(float) * kTb * kB);
  float*  hvA  = (float*)take(sizeof(float) * kTb * kB);
  float*  hvB  = (float*)take(sizeof(float) * kTb * kB);
  bf16_t* Wf   = (bf16_t*)take(sizeof(bf16_t) * (size_t)kKT * kNT * 512);
  bf16_t* Uf   = (bf16_t*)take(sizeof(bf16_t) * (size_t)kKT * kNT * 512);
  unsigned char* mBf  = (unsigned char*)take(kTb * kB);
  unsigned char* m2Bf = (unsigned char*)take(kTb * kB);

  k_prep_seq<<<(kTb * kB * kD) / 256, 256, 0, stream>>>(x, seqA);
  k_prep_scal<<<(kTb * kB + 255) / 256, 256, 0, stream>>>(mask, fkA, hvA, mBf, m2Bf);
  k_pack<<<(kKT * kNT * 512 + 255) / 256, 256, 0, stream>>>(W, Wf);
  k_pack<<<(kKT * kNT * 512 + 255) / 256, 256, 0, stream>>>(U, Uf);

  float* sin = seqA;  float* sout = seqB;
  float* fkp = fkA;   float* fkc = fkB;
  float* hvp = hvA;   float* hvc = hvB;

  // big GEMM: 2x3-tile wave blocks; 1024 M-tiles -> 512 M-groups, 11 N-groups
  const int bigMGroups = (kTb * kB) / 32;                  // 512
  const int bigWaves   = bigMGroups * (kNT / 3);           // 5632
  const int bigBlocks  = (bigWaves + 3) / 4;               // 4 waves / block
  // step GEMM: 1 tile/wave, 8 M-tiles x 33 N-tiles = 264 waves
  const int stepWaves  = (kB / 16) * kNT;
  const int stepBlocks = (stepWaves + 3) / 4;

  for (int d = 0; d < kDepth; ++d) {
    // S0 = ln(xv @ W, g0, b0) for all (t, b) — big parallel WMMA GEMM
    k_gemm<2, 3><<<bigBlocks, 128, 0, stream>>>(sin, Wf, S0, bigMGroups);
    k_ln_rows<<<kTb * kB, 256, 0, stream>>>(S0, g0, b0);
    // sequential time scan
    for (int t = 0; t < kTb; ++t) {
      if (t > 0)
        k_gemm<1, 1><<<stepBlocks, 128, 0, stream>>>(
            sout + (size_t)(t - 1) * kB * kD, Uf, S2, kB / 16);
      k_step_update<<<kB, 256, 0, stream>>>(S0, S2, sin, sout, fkp, hvp, fkc, hvc,
                                            mBf, m2Bf, bvec, g1, b1, t);
    }
    float* tmp;
    tmp = sin; sin = sout; sout = tmp;
    tmp = fkp; fkp = fkc; fkc = tmp;
    tmp = hvp; hvp = hvc; hvc = tmp;
  }
  k_final<<<(kB * kH + 255) / 256, 256, 0, stream>>>(sin, (float*)d_out);
}
